// Block_29678224015698
// MI455X (gfx1250) — matrix operand order repaired and checked by reading, not compiled, not run on hardware
//
#include <hip/hip_runtime.h>
#include <hip/hip_bf16.h>

typedef __bf16 bf16;
typedef __attribute__((ext_vector_type(16))) __bf16 v16bf;
typedef __attribute__((ext_vector_type(8)))  __bf16 v8bf;
typedef __attribute__((ext_vector_type(8)))  float  v8f;
typedef __attribute__((ext_vector_type(4)))  unsigned int v4u;
typedef __attribute__((ext_vector_type(8)))  int v8i;
typedef __attribute__((ext_vector_type(4)))  int v4i;

#define D_MODEL 1024
#define N_HEADS 16
#define HEAD_DIM 64
#define SEQ_T 2048
#define BATCH 2
#define M_TOK (BATCH * SEQ_T)   // 4096
#define FF_DIM (4 * D_MODEL)    // 4096
#define QKV_N (3 * D_MODEL)     // 3072

static __device__ __forceinline__ v16bf cat8(v8bf a, v8bf b) {
  return __builtin_shufflevector(a, b, 0,1,2,3,4,5,6,7,8,9,10,11,12,13,14,15);
}

static __device__ __forceinline__ void wmma_bf16(v16bf a, v16bf b, v8f& c) {
  c = __builtin_amdgcn_wmma_f32_16x16x32_bf16(false, a, false, b, (short)0, c, false, false);
}

// ---------------------------------------------------------------------------
// Elementwise f32 -> bf16 cast (weights)
// ---------------------------------------------------------------------------
__global__ void cast_bf16_kernel(const float* __restrict__ src, bf16* __restrict__ dst, int n) {
  int i = blockIdx.x * blockDim.x + threadIdx.x;
  if (i < n) dst[i] = (bf16)src[i];
}

// ---------------------------------------------------------------------------
// RMSNorm: one block per row of D_MODEL, output bf16
// ---------------------------------------------------------------------------
__global__ __launch_bounds__(256) void rmsnorm_kernel(const float* __restrict__ x,
                                                      const float* __restrict__ w,
                                                      bf16* __restrict__ out) {
  const int row = blockIdx.x;
  const float* xr = x + (size_t)row * D_MODEL;
  float s = 0.f;
  for (int i = threadIdx.x; i < D_MODEL; i += 256) { float v = xr[i]; s += v * v; }
  __shared__ float red[256];
  red[threadIdx.x] = s;
  __syncthreads();
  for (int off = 128; off > 0; off >>= 1) {
    if (threadIdx.x < off) red[threadIdx.x] += red[threadIdx.x + off];
    __syncthreads();
  }
  float inv = rsqrtf(red[0] * (1.0f / D_MODEL) + 1e-5f);
  for (int i = threadIdx.x; i < D_MODEL; i += 256)
    out[(size_t)row * D_MODEL + i] = (bf16)(xr[i] * inv * w[i]);
}

// ---------------------------------------------------------------------------
// Tiled WMMA GEMM with TDM-staged A block:
// C[M,N] = A[M,K] (bf16, row-major) x W[N,K]^T (bf16, row-major)
// Per block: A slice 128x32 staged into LDS by the Tensor Data Mover
// (double-buffered, issued by wave 0, synced via s_wait_tensorcnt + barrier).
// B fragments read directly from global (weights are L2-resident).
// MODE 0: outF = gemm
// MODE 1: outF = gemm + aux           (residual add)
// MODE 2: outB = bf16(silu(gemm)*aux) (SwiGLU gate fusion)
// Block = 256 thr = 8 waves (2x4), block tile 128x256, wave tile 64x64.
// ---------------------------------------------------------------------------
template<int MODE>
__global__ __launch_bounds__(256) void gemm_wmma_kernel(const bf16* __restrict__ A,
                                                        const bf16* __restrict__ W,
                                                        const float* __restrict__ aux,
                                                        float* __restrict__ outF,
                                                        bf16* __restrict__ outB,
                                                        int M, int N, int K) {
  const int lane = threadIdx.x & 31;
  const int wave = threadIdx.x >> 5;
  const int l15  = lane & 15;
  const int hi   = lane >> 4;
  const int m0 = blockIdx.x * 128 + (wave & 1) * 64;
  const int n0 = blockIdx.y * 256 + (wave >> 1) * 64;

  __shared__ __align__(16) bf16 sA[2][128 * 32];   // double-buffered A slice
  const unsigned ldsbase = (unsigned)(size_t)&sA[0][0];

  // TDM descriptor: 2D tile 32(contig, bf16) x 128 rows, row stride = K elems
  auto tdm_issue = [&](int buf, int k0) {
    unsigned long long ga =
        (unsigned long long)(size_t)(A + (size_t)blockIdx.x * 128 * K + k0);
    v4u g0;
    g0.x = 1u;                                            // count=1 (valid D#)
    g0.y = ldsbase + (unsigned)buf * (128u * 32u * 2u);   // lds_addr
    g0.z = (unsigned)ga;                                  // global_addr[31:0]
    g0.w = (unsigned)((ga >> 32) & 0x1FFFFFFull) | (2u << 30); // addr[56:32] | type=2
    v8i g1;
    g1[0] = 0x00010000;                                   // data_size=1 (2 bytes)
    g1[1] = (int)(((unsigned)K & 0xFFFFu) << 16);         // tensor_dim0 lo16
    g1[2] = (int)((((unsigned)K >> 16) & 0xFFFFu) | (128u << 16)); // dim0 hi | tensor_dim1 lo
    g1[3] = (int)(32u << 16);                             // tensor_dim1 hi=0 | tile_dim0=32
    g1[4] = 128;                                          // tile_dim1=128, tile_dim2=0
    g1[5] = K;                                            // tensor_dim0_stride lo32
    g1[6] = 0;
    g1[7] = 0;
    v4i gz4 = {0, 0, 0, 0};
    v8i gz8 = {0, 0, 0, 0, 0, 0, 0, 0};
    __builtin_amdgcn_tensor_load_to_lds(g0, g1, gz4, gz4, gz8, 0);
  };

  v8f acc[4][4];
#pragma unroll
  for (int i = 0; i < 4; ++i)
#pragma unroll
    for (int j = 0; j < 4; ++j) acc[i][j] = (v8f){0.f,0.f,0.f,0.f,0.f,0.f,0.f,0.f};

  if (wave == 0) {
    tdm_issue(0, 0);
    __builtin_amdgcn_s_wait_tensorcnt(0);
  }
  __syncthreads();

  const int arow = (wave & 1) * 64;
  for (int k0 = 0; k0 < K; k0 += 32) {
    const int cur = (k0 >> 5) & 1;
    const bool more = (k0 + 32) < K;
    if (wave == 0 && more) tdm_issue(cur ^ 1, k0 + 32);   // prefetch next slice

    v16bf afr[4];
#pragma unroll
    for (int mt = 0; mt < 4; ++mt) {
      const bf16* p = &sA[cur][(arow + mt * 16 + l15) * 32 + hi * 8];
      afr[mt] = cat8(*(const v8bf*)p, *(const v8bf*)(p + 16));
    }
#pragma unroll
    for (int nt = 0; nt < 4; ++nt) {
      const bf16* p = W + (size_t)(n0 + nt * 16 + l15) * K + k0 + hi * 8;
      v16bf bfr = cat8(*(const v8bf*)p, *(const v8bf*)(p + 16));
#pragma unroll
      for (int mt = 0; mt < 4; ++mt) wmma_bf16(afr[mt], bfr, acc[mt][nt]);
    }

    if (wave == 0 && more) __builtin_amdgcn_s_wait_tensorcnt(0);
    __syncthreads();
  }

#pragma unroll
  for (int mt = 0; mt < 4; ++mt)
#pragma unroll
    for (int nt = 0; nt < 4; ++nt)
#pragma unroll
      for (int r = 0; r < 8; ++r) {
        int m = m0 + mt * 16 + r + hi * 8;
        int n = n0 + nt * 16 + l15;
        size_t idx = (size_t)m * N + n;
        float v = acc[mt][nt][r];
        if (MODE == 0) {
          outF[idx] = v;
        } else if (MODE == 1) {
          outF[idx] = v + aux[idx];
        } else { // MODE == 2
          float sg = v / (1.0f + __expf(-v));
          outB[idx] = (bf16)(sg * aux[idx]);
        }
      }
}

// ---------------------------------------------------------------------------
// RoPE + head reshape. qkv f32 [B,T,3D] ->
//   Q bf16 [B*NH, T, 64] (scaled 1/sqrt(64)), K bf16 [B*NH, T, 64],
//   Vt bf16 [B*NH, 64, T] (transposed for contiguous P@V B-fragments)
// ---------------------------------------------------------------------------
__global__ __launch_bounds__(256) void rope_pack_kernel(const float* __restrict__ qkv,
                                                        bf16* __restrict__ Q,
                                                        bf16* __restrict__ Kh,
                                                        bf16* __restrict__ Vt) {
  int i = blockIdx.x * blockDim.x + threadIdx.x;   // 2^21 total
  int j = i & 31;
  int h = (i >> 5) & 15;
  int t = (i >> 9) & 2047;
  int b = i >> 20;
  const float* qp = qkv + ((size_t)(b * SEQ_T + t)) * QKV_N + h * HEAD_DIM;
  float theta = __expf(-((float)j * (1.0f / 32.0f)) * 9.210340371976184f); // 10000^-(j/32)
  float ang = (float)t * theta;
  float c = __cosf(ang), s = __sinf(ang);
  float q1 = qp[j],             q2 = qp[j + 32];
  float k1 = qp[D_MODEL + j],   k2 = qp[D_MODEL + j + 32];
  float v1 = qp[2*D_MODEL + j], v2 = qp[2*D_MODEL + j + 32];
  size_t hb = (size_t)(b * N_HEADS + h) * SEQ_T;
  size_t qi = (hb + t) * HEAD_DIM;
  Q[qi + j]      = (bf16)((q1 * c - q2 * s) * 0.125f);
  Q[qi + j + 32] = (bf16)((q1 * s + q2 * c) * 0.125f);
  Kh[qi + j]      = (bf16)(k1 * c - k2 * s);
  Kh[qi + j + 32] = (bf16)(k1 * s + k2 * c);
  size_t vb = (size_t)(b * N_HEADS + h) * HEAD_DIM * SEQ_T;
  Vt[vb + (size_t)j * SEQ_T + t]        = (bf16)v1;
  Vt[vb + (size_t)(j + 32) * SEQ_T + t] = (bf16)v2;
}

// ---------------------------------------------------------------------------
// Causal flash attention. grid (B*NH, T/64), 128 threads = 4 waves.
// Each wave owns 16 queries; per 64-key chunk: S = Q@K^T (WMMA), online
// softmax (LDS round-trip + shfl_xor(16) row reductions), O += P@V (WMMA).
// ---------------------------------------------------------------------------
__global__ __launch_bounds__(128) void flash_attn_kernel(const bf16* __restrict__ Q,
                                                         const bf16* __restrict__ Kh,
                                                         const bf16* __restrict__ Vt,
                                                         bf16* __restrict__ Y) {
  const int bh   = blockIdx.x;
  const int qc   = blockIdx.y;
  const int lane = threadIdx.x & 31;
  const int wave = threadIdx.x >> 5;
  const int l15  = lane & 15;
  const int hi   = lane >> 4;
  const int q0   = qc * 64 + wave * 16;

  const bf16* Qp = Q  + (size_t)bh * SEQ_T * HEAD_DIM;
  const bf16* Kp = Kh + (size_t)bh * SEQ_T * HEAD_DIM;
  const bf16* Vp = Vt + (size_t)bh * HEAD_DIM * SEQ_T;

  __shared__ __align__(16) float sS[4][16 * 64];
  __shared__ __align__(16) bf16  sP[4][16 * 64];
  __shared__ float sA[4][16];
  __shared__ float sL[4][16];

  // preload Q fragments (16 x 64 -> two 16x32 A tiles)
  v16bf qfr[2];
#pragma unroll
  for (int kt = 0; kt < 2; ++kt) {
    const bf16* p = Qp + (size_t)(q0 + l15) * HEAD_DIM + kt * 32 + hi * 8;
    qfr[kt] = cat8(*(const v8bf*)p, *(const v8bf*)(p + 16));
  }

  v8f o[4];
#pragma unroll
  for (int nt = 0; nt < 4; ++nt) o[nt] = (v8f){0.f,0.f,0.f,0.f,0.f,0.f,0.f,0.f};
  float m_i = -3.0e38f, l_i = 0.f;   // valid in lanes 0..15 (row = l15)

  const int nchunks = qc + 1;
  for (int kc = 0; kc < nchunks; ++kc) {
    const int kbase = kc * 64;
    v8f sacc[4];
#pragma unroll
    for (int nt = 0; nt < 4; ++nt) sacc[nt] = (v8f){0.f,0.f,0.f,0.f,0.f,0.f,0.f,0.f};
#pragma unroll
    for (int kt = 0; kt < 2; ++kt)
#pragma unroll
      for (int nt = 0; nt < 4; ++nt) {
        const bf16* p = Kp + (size_t)(kbase + nt * 16 + l15) * HEAD_DIM + kt * 32 + hi * 8;
        v16bf kfr = cat8(*(const v8bf*)p, *(const v8bf*)(p + 16));
        wmma_bf16(qfr[kt], kfr, sacc[nt]);
      }
    // causal mask + stash S to LDS (row-major 16x64 per wave)
#pragma unroll
    for (int nt = 0; nt < 4; ++nt)
#pragma unroll
      for (int r = 0; r < 8; ++r) {
        int qrow = r + hi * 8;
        int kidx = kbase + nt * 16 + l15;
        float v = sacc[nt][r];
        if (kidx > q0 + qrow) v = -1.0e30f;
        sS[wave][qrow * 64 + nt * 16 + l15] = v;
      }
    __syncthreads();
    // online softmax: lane L handles row L&15, columns [hi*32, hi*32+32);
    // pair lanes (L, L^16) combine via shfl_xor(16)
    {
      const float* row = &sS[wave][l15 * 64] + hi * 32;
      float mc = -3.0e38f;
#pragma unroll
      for (int j = 0; j < 32; ++j) mc = fmaxf(mc, row[j]);
      mc = fmaxf(mc, __shfl_xor(mc, 16, 32));
      // both pair lanes now agree on chunk max; combine with running max
      // (m_i only meaningful in hi==0 lane; broadcast result from hi==0)
      float nm = fmaxf(m_i, mc);
      float nm0 = hi ? __shfl_xor(nm, 16, 32) : nm;
      float ps = 0.f;
      bf16* prow = &sP[wave][l15 * 64] + hi * 32;
#pragma unroll
      for (int j = 0; j < 32; ++j) {
        float p = __expf(row[j] - nm0);
        prow[j] = (bf16)p;
        ps += p;
      }
      ps += __shfl_xor(ps, 16, 32);
      if (hi == 0) {
        float alpha = __expf(m_i - nm0);
        l_i = alpha * l_i + ps;
        m_i = nm0;
        sA[wave][l15] = alpha;
      }
    }
    __syncthreads();
    // rescale running output
#pragma unroll
    for (int nt = 0; nt < 4; ++nt)
#pragma unroll
      for (int r = 0; r < 8; ++r) o[nt][r] *= sA[wave][r + hi * 8];
    // O += P @ V  (P from LDS, V^T from global: both contiguous b128 frags)
#pragma unroll
    for (int kt = 0; kt < 2; ++kt) {
      const bf16* pp = &sP[wave][l15 * 64 + kt * 32 + hi * 8];
      v16bf pfr = cat8(*(const v8bf*)pp, *(const v8bf*)(pp + 16));
#pragma unroll
      for (int nt = 0; nt < 4; ++nt) {
        const bf16* vp = Vp + (size_t)(nt * 16 + l15) * SEQ_T + kbase + kt * 32 + hi * 8;
        v16bf vfr = cat8(*(const v8bf*)vp, *(const v8bf*)(vp + 16));
        wmma_bf16(pfr, vfr, o[nt]);
      }
    }
    __syncthreads();
  }

  if (hi == 0) sL[wave][l15] = l_i;
  __syncthreads();
  const int b = bh >> 4, h = bh & 15;
#pragma unroll
  for (int nt = 0; nt < 4; ++nt)
#pragma unroll
    for (int r = 0; r < 8; ++r) {
      int row = r + hi * 8;
      float val = o[nt][r] / sL[wave][row];
      size_t t = (size_t)(q0 + row);
      Y[((size_t)b * SEQ_T + t) * D_MODEL + h * HEAD_DIM + nt * 16 + l15] = (bf16)val;
    }
}

// ---------------------------------------------------------------------------
// Launch
// ---------------------------------------------------------------------------
static inline size_t align_up(size_t v, size_t a) { return (v + a - 1) & ~(a - 1); }

extern "C" void kernel_launch(void* const* d_in, const int* in_sizes, int n_in,
                              void* d_out, int out_size, void* d_ws, size_t ws_size,
                              hipStream_t stream) {
  (void)in_sizes; (void)n_in; (void)out_size; (void)ws_size;
  const float* x     = (const float*)d_in[0];
  const float* n1_w  = (const float*)d_in[1];
  const float* n2_w  = (const float*)d_in[2];
  const float* qkv_w = (const float*)d_in[3];
  const float* o_w   = (const float*)d_in[4];
  const float* w1_w  = (const float*)d_in[5];
  const float* wg_w  = (const float*)d_in[6];
  const float* w2_w  = (const float*)d_in[7];
  float* out = (float*)d_out;

  // workspace carve-up
  char* ws = (char*)d_ws;
  size_t off = 0;
  auto alloc = [&](size_t bytes) { void* p = ws + off; off = align_up(off + bytes, 256); return p; };
  bf16*  wqkv_b = (bf16*)alloc((size_t)QKV_N * D_MODEL * 2);
  bf16*  wo_b   = (bf16*)alloc((size_t)D_MODEL * D_MODEL * 2);
  bf16*  w1_b   = (bf16*)alloc((size_t)FF_DIM * D_MODEL * 2);
  bf16*  wg_b   = (bf16*)alloc((size_t)FF_DIM * D_MODEL * 2);
  bf16*  w2_b   = (bf16*)alloc((size_t)D_MODEL * FF_DIM * 2);
  bf16*  xn1    = (bf16*)alloc((size_t)M_TOK * D_MODEL * 2);
  float* qkv    = (float*)alloc((size_t)M_TOK * QKV_N * 4);
  bf16*  Qb     = (bf16*)alloc((size_t)M_TOK * D_MODEL * 2);
  bf16*  Kb     = (bf16*)alloc((size_t)M_TOK * D_MODEL * 2);
  bf16*  Vtb    = (bf16*)alloc((size_t)M_TOK * D_MODEL * 2);
  bf16*  Yb     = (bf16*)alloc((size_t)M_TOK * D_MODEL * 2);
  float* xattn  = (float*)alloc((size_t)M_TOK * D_MODEL * 4);
  bf16*  xn2    = (bf16*)alloc((size_t)M_TOK * D_MODEL * 2);
  float* h1     = (float*)alloc((size_t)M_TOK * FF_DIM * 4);
  bf16*  act    = (bf16*)alloc((size_t)M_TOK * FF_DIM * 2);

  auto cast = [&](const float* s, bf16* d, int n) {
    cast_bf16_kernel<<<(n + 255) / 256, 256, 0, stream>>>(s, d, n);
  };
  cast(qkv_w, wqkv_b, QKV_N * D_MODEL);
  cast(o_w,   wo_b,   D_MODEL * D_MODEL);
  cast(w1_w,  w1_b,   FF_DIM * D_MODEL);
  cast(wg_w,  wg_b,   FF_DIM * D_MODEL);
  cast(w2_w,  w2_b,   D_MODEL * FF_DIM);

  // x -> rmsnorm1 -> xn1 (bf16)
  rmsnorm_kernel<<<M_TOK, 256, 0, stream>>>(x, n1_w, xn1);

  // qkv = xn1 @ qkv_w^T  (f32)
  gemm_wmma_kernel<0><<<dim3(M_TOK / 128, QKV_N / 256), 256, 0, stream>>>(
      xn1, wqkv_b, nullptr, qkv, nullptr, M_TOK, QKV_N, D_MODEL);

  // rope + reshape -> Q, K, V^T (bf16)
  rope_pack_kernel<<<(BATCH * SEQ_T * N_HEADS * 32) / 256, 256, 0, stream>>>(qkv, Qb, Kb, Vtb);

  // causal attention -> Yb (bf16 [B*T, D])
  flash_attn_kernel<<<dim3(BATCH * N_HEADS, SEQ_T / 64), 128, 0, stream>>>(Qb, Kb, Vtb, Yb);

  // xattn = Yb @ o_w^T + x
  gemm_wmma_kernel<1><<<dim3(M_TOK / 128, D_MODEL / 256), 256, 0, stream>>>(
      Yb, wo_b, x, xattn, nullptr, M_TOK, D_MODEL, D_MODEL);

  // rmsnorm2 -> xn2 (bf16)
  rmsnorm_kernel<<<M_TOK, 256, 0, stream>>>(xattn, n2_w, xn2);

  // h1 = xn2 @ w1^T (f32)
  gemm_wmma_kernel<0><<<dim3(M_TOK / 128, FF_DIM / 256), 256, 0, stream>>>(
      xn2, w1_b, nullptr, h1, nullptr, M_TOK, FF_DIM, D_MODEL);

  // act = silu(xn2 @ wg^T) * h1  (bf16)
  gemm_wmma_kernel<2><<<dim3(M_TOK / 128, FF_DIM / 256), 256, 0, stream>>>(
      xn2, wg_b, h1, nullptr, act, M_TOK, FF_DIM, D_MODEL);

  // out = act @ w2^T + xattn
  gemm_wmma_kernel<1><<<dim3(M_TOK / 128, D_MODEL / 256), 256, 0, stream>>>(
      act, w2_b, xattn, out, nullptr, M_TOK, D_MODEL, FF_DIM);
}
